// IAMGRUPolicy_dynamic_42399917146384
// MI455X (gfx1250) — compile-verified, matrix operand order checked by reading, not run on hardware
//
#include <hip/hip_runtime.h>
#include <hip/hip_bf16.h>

typedef __attribute__((ext_vector_type(16))) _Float16 v16h;
typedef __attribute__((ext_vector_type(8)))  _Float16 v8h;
typedef __attribute__((ext_vector_type(8)))  float    v8f;
typedef __attribute__((ext_vector_type(4)))  int      v4i_;

#define NW_  2048
#define OBS_ 512
#define HS_  512
#define HS2_ 512
#define HM_  256
#define CTX_ 128
#define ACT_ 18

#if __has_builtin(__builtin_amdgcn_global_load_async_to_lds_b128)
#define HAVE_ASYNC_LDS 1
#else
#define HAVE_ASYNC_LDS 0
#endif

typedef __attribute__((address_space(1))) v4i_* as1_v4i;
typedef __attribute__((address_space(3))) v4i_* as3_v4i;

// ---------------------------------------------------------------------------
// f32 -> f16 conversion kernels
// ---------------------------------------------------------------------------
__global__ void cvt_f32_f16_kernel(const float* __restrict__ src,
                                   _Float16* __restrict__ dst, int n) {
    int i = blockIdx.x * blockDim.x + threadIdx.x;
    if (i < n) dst[i] = (_Float16)src[i];
}

// Extract Wc[:, 1:] (shape 128 x 256 from row stride 257) into f16
__global__ void wc_extract_kernel(const float* __restrict__ Wc,
                                  _Float16* __restrict__ dst) {
    int i = blockIdx.x * blockDim.x + threadIdx.x; // 128*256
    int c = i >> 8;
    int k = i & 255;
    dst[i] = (_Float16)Wc[c * 257 + 1 + k];
}

// ---------------------------------------------------------------------------
// Async 16-byte stage into LDS (CDNA5 GLOBAL_LOAD_ASYNC_TO_LDS_B128, ASYNCcnt)
// with a synchronous fallback if the builtin is unavailable.
// ---------------------------------------------------------------------------
__device__ __forceinline__ void stage16(const _Float16* g, _Float16* l) {
#if HAVE_ASYNC_LDS
    __builtin_amdgcn_global_load_async_to_lds_b128(
        (as1_v4i)(void*)g, (as3_v4i)(void*)l, 0, 0);
#else
    *(v8h*)l = *(const v8h*)g;
#endif
}

__device__ __forceinline__ void wait_stage() {
#if HAVE_ASYNC_LDS
#if __has_builtin(__builtin_amdgcn_s_wait_asynccnt)
    __builtin_amdgcn_s_wait_asynccnt(0);
#else
    asm volatile("s_wait_asynccnt 0x0" ::: "memory");
#endif
#endif
}

// ---------------------------------------------------------------------------
// WMMA GEMM:  C[M,N] = act( A[M,K] * Bw[N,K]^T + bias[N] )
// A, Bw f16 row-major. Block = 256 threads (8 waves). Wave wv computes the
// 16x64 strip rows = blockIdx.y*128 + wv*16, cols = blockIdx.x*64.
// The 64x32 B tile (shared by all 8 waves) is double-buffered in LDS via
// async-to-LDS DMA; rows padded to 40 halves (80B) -> conflict-free
// ds_load_b128 fragment reads (20-bank row stride).
// A-fragment lane layout per CDNA5 ISA (16-bit A 16x32):
//   lane L: row M = L%16, element j -> K = k0 + (j/8)*16 + (L/16)*8 + (j%8)
// ---------------------------------------------------------------------------
#define LDS_ROW_H 40

template<bool RELU, bool HALF_OUT>
__global__ void gemm_wmma_kernel(const _Float16* __restrict__ A,
                                 const _Float16* __restrict__ Bw,
                                 const float* __restrict__ bias,
                                 void* __restrict__ Cout,
                                 int K, int ldc, int col_off) {
    __shared__ _Float16 bsm[2][64 * LDS_ROW_H];

    const int tid  = threadIdx.x;
    const int lane = tid & 31;
    const int wv   = tid >> 5;
    const int rowBase = blockIdx.y * 128 + wv * 16;
    const int colBase = blockIdx.x * 64;
    const int r16   = lane & 15;
    const int klane = (lane >> 4) * 8;

    // cooperative staging: thread tid copies 16B: row = tid/4, chunk = tid%4
    const int srow   = tid >> 2;
    const int schunk = (tid & 3) * 8;
    const _Float16* gsrc = Bw + (size_t)(colBase + srow) * K + schunk;
    _Float16* ldst[2] = { &bsm[0][srow * LDS_ROW_H + schunk],
                          &bsm[1][srow * LDS_ROW_H + schunk] };

    // prologue: stage tile k0 = 0 into buffer 0
    stage16(gsrc, ldst[0]);
    wait_stage();
    __syncthreads();

    v8f acc[4] = {};
    const _Float16* arow = A + (size_t)(rowBase + r16) * K + klane;

    int cur = 0;
    for (int k0 = 0; k0 < K; k0 += 32) {
        const int k1 = k0 + 32;
        if (k1 < K) stage16(gsrc + k1, ldst[cur ^ 1]);   // prefetch next tile

        union { v16h v; v8h h[2]; } au;
        au.h[0] = *(const v8h*)(arow + k0);
        au.h[1] = *(const v8h*)(arow + k0 + 16);

        const _Float16* bbase = bsm[cur];
#pragma unroll
        for (int t = 0; t < 4; ++t) {
            const _Float16* br = bbase + (t * 16 + r16) * LDS_ROW_H + klane;
            union { v16h v; v8h h[2]; } bu;
            bu.h[0] = *(const v8h*)br;          // ds_load_b128
            bu.h[1] = *(const v8h*)(br + 16);   // ds_load_b128
            acc[t] = __builtin_amdgcn_wmma_f32_16x16x32_f16(
                false, au.v, false, bu.v, (short)0, acc[t], false, false);
        }
        wait_stage();
        __syncthreads();
        cur ^= 1;
    }

    const int hi = lane >> 4;
#pragma unroll
    for (int t = 0; t < 4; ++t) {
        int col = colBase + t * 16 + r16;
        float bv = bias ? bias[col] : 0.0f;
#pragma unroll
        for (int j = 0; j < 8; ++j) {
            int row = rowBase + j + hi * 8;
            float v = acc[t][j] + bv;
            if (RELU) v = fmaxf(v, 0.0f);
            if (HALF_OUT)
                ((_Float16*)Cout)[(size_t)row * ldc + col_off + col] = (_Float16)v;
            else
                ((float*)Cout)[(size_t)row * ldc + col_off + col] = v;
        }
    }
}

// ---------------------------------------------------------------------------
// Attention: per batch row b:
//   alog[o] = sum_c Wa[c] * tanh(obs[b,o]*Wc[c,0] + hterm[b,c]) + ba
//   attn = softmax_o(alog);  x[b] = sum_o attn[o]*obs[b,o]
// One 256-thread block per b; each thread owns o = t and o = t+256.
// ---------------------------------------------------------------------------
__global__ void attn_kernel(const float* __restrict__ obs,
                            const float* __restrict__ hterm,
                            const float* __restrict__ Wc,
                            const float* __restrict__ Wa,
                            const float* __restrict__ ba,
                            float* __restrict__ xout) {
    __shared__ float ht[CTX_], w0s[CTX_], was[CTX_];
    __shared__ float red[256];
    const int b = blockIdx.x, t = threadIdx.x;

    if (t < CTX_) {
        ht[t]  = hterm[b * CTX_ + t];
        w0s[t] = Wc[t * (HM_ + 1)];
        was[t] = Wa[t];
    }
    __syncthreads();

    const float ov1 = obs[b * OBS_ + t];
    const float ov2 = obs[b * OBS_ + 256 + t];
    float a1 = 0.0f, a2 = 0.0f;
#pragma unroll 4
    for (int c = 0; c < CTX_; ++c) {
        float w = w0s[c], h = ht[c], wa = was[c];
        a1 = fmaf(wa, tanhf(fmaf(ov1, w, h)), a1);
        a2 = fmaf(wa, tanhf(fmaf(ov2, w, h)), a2);
    }
    const float bav = ba[0];
    a1 += bav; a2 += bav;

    // block max
    red[t] = fmaxf(a1, a2);
    __syncthreads();
    for (int s = 128; s >= 1; s >>= 1) {
        if (t < s) red[t] = fmaxf(red[t], red[t + s]);
        __syncthreads();
    }
    const float mx = red[0];
    __syncthreads();

    const float e1 = expf(a1 - mx), e2 = expf(a2 - mx);
    red[t] = e1 + e2;
    __syncthreads();
    for (int s = 128; s >= 1; s >>= 1) {
        if (t < s) red[t] += red[t + s];
        __syncthreads();
    }
    const float se = red[0];
    __syncthreads();

    red[t] = e1 * ov1 + e2 * ov2;
    __syncthreads();
    for (int s = 128; s >= 1; s >>= 1) {
        if (t < s) red[t] += red[t + s];
        __syncthreads();
    }
    if (t == 0) xout[b] = red[0] / se;
}

// ---------------------------------------------------------------------------
// GRU cell elementwise; writes hnew (fp16) into concat buffer cols [512,768)
// gh already holds hp @ W_hh^T + b_hh (f32).
// ---------------------------------------------------------------------------
__device__ __forceinline__ float sigmoidf_(float x) {
    return 1.0f / (1.0f + expf(-x));
}

__global__ void gru_kernel(const float* __restrict__ x,
                           const float* __restrict__ W_ih,
                           const float* __restrict__ b_ih,
                           const float* __restrict__ gh,
                           const float* __restrict__ hp,
                           _Float16* __restrict__ concat_h) {
    int idx = blockIdx.x * blockDim.x + threadIdx.x; // NW_*HM_
    int b = idx >> 8;
    int j = idx & 255;
    float xb = x[b];
    float ir  = fmaf(xb, W_ih[j],           b_ih[j]);
    float iz  = fmaf(xb, W_ih[HM_ + j],     b_ih[HM_ + j]);
    float inn = fmaf(xb, W_ih[2 * HM_ + j], b_ih[2 * HM_ + j]);
    const float* g = gh + (size_t)b * (3 * HM_);
    float hr = g[j], hz = g[HM_ + j], hn = g[2 * HM_ + j];
    float r = sigmoidf_(ir + hr);
    float z = sigmoidf_(iz + hz);
    float n = tanhf(fmaf(r, hn, inn));
    float h = hp[(size_t)b * HM_ + j];
    float hnew = (1.0f - z) * n + z * h;
    concat_h[(size_t)b * (HS_ + HM_) + HS_ + j] = (_Float16)hnew;
}

// ---------------------------------------------------------------------------
// Head: logits(18) + value(1) from out[b,:512]; threefry2x32 Gumbel argmax
// (JAX key(42) -> (0,42); counts split-in-half convention), log_softmax.
// d_out layout: action[2048] | value[2048] | log_prob[2048]  (all as float)
// ---------------------------------------------------------------------------
__device__ __forceinline__ unsigned int rotl32_(unsigned int x, int r) {
    return (x << r) | (x >> (32 - r));
}

__device__ unsigned int jax_threefry_bits(unsigned int flat, unsigned int half) {
    unsigned int p = (flat < half) ? flat : (flat - half);
    unsigned int x0 = p, x1 = p + half;
    const unsigned int ks0 = 0u, ks1 = 42u;
    const unsigned int ks2 = 0x1BD11BDAu ^ ks0 ^ ks1;
    const int ra[4] = {13, 15, 26, 6};
    const int rb[4] = {17, 29, 16, 24};
    x0 += ks0; x1 += ks1;
    for (int i = 0; i < 4; ++i) { x0 += x1; x1 = rotl32_(x1, ra[i]); x1 ^= x0; }
    x0 += ks1; x1 += ks2 + 1u;
    for (int i = 0; i < 4; ++i) { x0 += x1; x1 = rotl32_(x1, rb[i]); x1 ^= x0; }
    x0 += ks2; x1 += ks0 + 2u;
    for (int i = 0; i < 4; ++i) { x0 += x1; x1 = rotl32_(x1, ra[i]); x1 ^= x0; }
    x0 += ks0; x1 += ks1 + 3u;
    for (int i = 0; i < 4; ++i) { x0 += x1; x1 = rotl32_(x1, rb[i]); x1 ^= x0; }
    x0 += ks1; x1 += ks2 + 4u;
    for (int i = 0; i < 4; ++i) { x0 += x1; x1 = rotl32_(x1, ra[i]); x1 ^= x0; }
    x0 += ks2; x1 += ks0 + 5u;
    return (flat < half) ? x0 : x1;
}

__global__ void head_kernel(const float* __restrict__ outb,
                            const float* __restrict__ Wact,
                            const float* __restrict__ bact,
                            const float* __restrict__ Wcrit,
                            const float* __restrict__ bcrit,
                            float* __restrict__ d_out) {
    __shared__ float so[HS2_];
    __shared__ float logits[ACT_ + 1];
    const int b = blockIdx.x, t = threadIdx.x;

    so[t]       = outb[(size_t)b * HS2_ + t];
    so[t + 256] = outb[(size_t)b * HS2_ + 256 + t];
    __syncthreads();

    if (t <= ACT_) {
        const float* w = (t < ACT_) ? (Wact + (size_t)t * HS2_) : Wcrit;
        float acc = 0.0f;
        for (int k = 0; k < HS2_; ++k) acc = fmaf(so[k], w[k], acc);
        acc += (t < ACT_) ? bact[t] : bcrit[0];
        logits[t] = acc;
    }
    __syncthreads();

    if (t == 0) {
        const unsigned int half = (unsigned)(NW_ * ACT_) / 2u;
        float best = -3.4e38f;
        int bi = 0;
        float mx = -3.4e38f;
#pragma unroll
        for (int j = 0; j < ACT_; ++j) mx = fmaxf(mx, logits[j]);
        float s = 0.0f;
        for (int j = 0; j < ACT_; ++j) {
            unsigned int bits = jax_threefry_bits((unsigned)(b * ACT_ + j), half);
            float u = (float)(bits >> 9) * (1.0f / 8388608.0f); // [0,1)
            u = fmaxf(u, 1.1754944e-38f);
            float gum = -logf(-logf(u));
            float v = logits[j] + gum;
            if (v > best) { best = v; bi = j; }
            s += expf(logits[j] - mx);
        }
        float lse = mx + logf(s);
        d_out[b]            = (float)bi;          // action
        d_out[NW_ + b]      = logits[ACT_];       // value
        d_out[2 * NW_ + b]  = logits[bi] - lse;   // log_prob
    }
}

// ---------------------------------------------------------------------------
extern "C" void kernel_launch(void* const* d_in, const int* in_sizes, int n_in,
                              void* d_out, int out_size, void* d_ws, size_t ws_size,
                              hipStream_t stream) {
    (void)in_sizes; (void)n_in; (void)out_size; (void)ws_size;
    const float* obs   = (const float*)d_in[0];
    const float* hp    = (const float*)d_in[1];   // (1,2048,256) contiguous
    const float* W1    = (const float*)d_in[2];
    const float* b1    = (const float*)d_in[3];
    const float* Wc    = (const float*)d_in[4];
    const float* bc    = (const float*)d_in[5];
    const float* Wa    = (const float*)d_in[6];
    const float* ba    = (const float*)d_in[7];
    const float* W_ih  = (const float*)d_in[8];
    const float* b_ih  = (const float*)d_in[9];
    const float* W_hh  = (const float*)d_in[10];
    const float* b_hh  = (const float*)d_in[11];
    const float* W2    = (const float*)d_in[12];
    const float* b2    = (const float*)d_in[13];
    const float* Wact  = (const float*)d_in[14];
    const float* bact  = (const float*)d_in[15];
    const float* Wcrit = (const float*)d_in[16];
    const float* bcrit = (const float*)d_in[17];

    char* ws = (char*)d_ws;
    size_t off = 0;
    auto alloc = [&](size_t bytes) -> void* {
        void* p = ws + off;
        off += (bytes + 255) & ~(size_t)255;
        return p;
    };

    _Float16* obs_h   = (_Float16*)alloc((size_t)NW_ * OBS_ * 2);
    _Float16* w1_h    = (_Float16*)alloc((size_t)HS_ * OBS_ * 2);
    _Float16* hp_h    = (_Float16*)alloc((size_t)NW_ * HM_ * 2);
    _Float16* wc_h    = (_Float16*)alloc((size_t)CTX_ * HM_ * 2);
    _Float16* whh_h   = (_Float16*)alloc((size_t)3 * HM_ * HM_ * 2);
    _Float16* w2_h    = (_Float16*)alloc((size_t)HS2_ * (HS_ + HM_) * 2);
    _Float16* concat_h= (_Float16*)alloc((size_t)NW_ * (HS_ + HM_) * 2);
    float*    hterm   = (float*)alloc((size_t)NW_ * CTX_ * 4);
    float*    xbuf    = (float*)alloc((size_t)NW_ * 4);
    float*    gh      = (float*)alloc((size_t)NW_ * 3 * HM_ * 4);
    float*    outbuf  = (float*)alloc((size_t)NW_ * HS2_ * 4);

    // convert inputs/weights to f16
    auto cvt = [&](const float* s, _Float16* d, int n) {
        cvt_f32_f16_kernel<<<(n + 255) / 256, 256, 0, stream>>>(s, d, n);
    };
    cvt(obs,  obs_h, NW_ * OBS_);
    cvt(W1,   w1_h,  HS_ * OBS_);
    cvt(hp,   hp_h,  NW_ * HM_);
    cvt(W_hh, whh_h, 3 * HM_ * HM_);
    cvt(W2,   w2_h,  HS2_ * (HS_ + HM_));
    wc_extract_kernel<<<(CTX_ * HM_ + 255) / 256, 256, 0, stream>>>(Wc, wc_h);

    // feat = relu(obs @ W1^T + b1) -> concat_h[:, 0:512] (fp16)
    gemm_wmma_kernel<true, true><<<dim3(HS_ / 64, NW_ / 128), 256, 0, stream>>>(
        obs_h, w1_h, b1, concat_h, OBS_, HS_ + HM_, 0);

    // h_term = hp @ Wc[:,1:]^T + bc (f32)
    gemm_wmma_kernel<false, false><<<dim3(CTX_ / 64, NW_ / 128), 256, 0, stream>>>(
        hp_h, wc_h, bc, hterm, HM_, CTX_, 0);

    // attention -> x
    attn_kernel<<<NW_, 256, 0, stream>>>(obs, hterm, Wc, Wa, ba, xbuf);

    // gh = hp @ W_hh^T + b_hh (f32)
    gemm_wmma_kernel<false, false><<<dim3(3 * HM_ / 64, NW_ / 128), 256, 0, stream>>>(
        hp_h, whh_h, b_hh, gh, HM_, 3 * HM_, 0);

    // GRU -> concat_h[:, 512:768] (fp16)
    gru_kernel<<<(NW_ * HM_) / 256, 256, 0, stream>>>(xbuf, W_ih, b_ih, gh, hp, concat_h);

    // out = relu(concat @ W2^T + b2) (f32)
    gemm_wmma_kernel<true, false><<<dim3(HS2_ / 64, NW_ / 128), 256, 0, stream>>>(
        concat_h, w2_h, b2, outbuf, HS_ + HM_, HS2_, 0);

    // head: logits, sampling, value, log_prob
    head_kernel<<<NW_, 256, 0, stream>>>(outbuf, Wact, bact, Wcrit, bcrit, (float*)d_out);
}
